// GraphConv_59837484368264
// MI455X (gfx1250) — compile-verified, matrix operand order checked by reading, not run on hardware
//
#include <hip/hip_runtime.h>
#include <hip/hip_bf16.h>

typedef __attribute__((ext_vector_type(16))) _Float16 v16h;
typedef __attribute__((ext_vector_type(8)))  _Float16 v8h;
typedef __attribute__((ext_vector_type(4)))  _Float16 v4h;
typedef __attribute__((ext_vector_type(8)))  float    v8f;

#define DEVINL __device__ __forceinline__

constexpr int NUSR  = 100000;
constexpr int NITM  = 50000;
constexpr int NENT  = 80000;
constexpr int NN    = NUSR + NENT;     // 180000 nodes
constexpr int NEDGE = 800000;
constexpr int D     = 64;
constexpr int NSLOT = 2 * NUSR + NENT; // 280000 offset slots
constexpr int TS    = 72;              // LDS tile row stride (halves), conflict-free

// ---- ordered-uint encoding for float atomic min/max ----
DEVINL unsigned fkey(float f) {
  int i = __float_as_int(f);
  return (unsigned)(i ^ ((i >> 31) | 0x80000000));
}
DEVINL float fdecode(unsigned k) {
  int i = (k & 0x80000000u) ? (int)(k ^ 0x80000000u) : ~(int)k;
  return __int_as_float(i);
}

DEVINL v8f wmma16(v16h a, v16h b, v8f c) {
  // D = A(16x32 f16) x B(32x16 f16) + C(16x16 f32)
  return __builtin_amdgcn_wmma_f32_16x16x32_f16(false, a, false, b, (short)0, c,
                                                false, false);
}

DEVINL void lds_fence_wave() {
  __builtin_amdgcn_wave_barrier();
  asm volatile("s_wait_dscnt 0" ::: "memory");
  __builtin_amdgcn_wave_barrier();
}

// A fragment (16x32 f16) from an LDS tile with row stride TS halves.
// lane l: row m=l&15, K-base kb=(l>>4)*8; halves 0..7 -> K=kc*32+kb..+7,
// halves 8..15 -> K=kc*32+16+kb..+7  (per ISA 16-bit A layout).
DEVINL v16h ldsA(const _Float16* t, int lane, int kc) {
  int m  = lane & 15;
  int kb = (lane >> 4) << 3;
  v8h lo = *(const v8h*)(t + m * TS + kc * 32 + kb);
  v8h hi = *(const v8h*)(t + m * TS + kc * 32 + 16 + kb);
  return __builtin_shufflevector(lo, hi, 0, 1, 2, 3, 4, 5, 6, 7, 8, 9, 10, 11,
                                 12, 13, 14, 15);
}
// Pre-swizzled weight fragment: contiguous 16 halves per (nb,kc,lane).
DEVINL v16h ldsW(const _Float16* w, int nb, int kc, int lane) {
  return *(const v16h*)(w + (((nb << 1) | kc) * 32 + lane) * 16);
}

// 16x64 = 16x64 @ 64x64^T + bias, via 8 WMMAs. acc[nb] covers out cols nb*16..+15.
DEVINL void gemm4(const _Float16* aT, const _Float16* wT, const float bn[4],
                  int lane, v8f acc[4]) {
  v16h a0 = ldsA(aT, lane, 0);
  v16h a1 = ldsA(aT, lane, 1);
#pragma unroll
  for (int nb = 0; nb < 4; nb++) {
    float b = bn[nb];
    v8f c = {b, b, b, b, b, b, b, b};
    c = wmma16(a0, ldsW(wT, nb, 0, lane), c);
    c = wmma16(a1, ldsW(wT, nb, 1, lane), c);
    acc[nb] = c;
  }
}

// Gather 16 indexed f32 rows -> f16 LDS tile. lane covers (row=lane/2, 32 cols).
DEVINL void gather16_idx(_Float16* tile, const float* base, const int* rows,
                         int lane) {
  int r = lane >> 1;
  int c0 = (lane & 1) << 5;
  int t = rows[r];
  const float4* p = (const float4*)(base + (size_t)t * D + c0);
  _Float16* d = tile + r * TS + c0;
#pragma unroll
  for (int q = 0; q < 8; q++) {
    float4 v = p[q];
    v4h h = {(_Float16)v.x, (_Float16)v.y, (_Float16)v.z, (_Float16)v.w};
    *(v4h*)(d + q * 4) = h;
  }
}
// Gather 16 consecutive rows starting at base.
DEVINL void gather16_seq(_Float16* tile, const float* base, int lane) {
  int r = lane >> 1;
  int c0 = (lane & 1) << 5;
  const float4* p = (const float4*)(base + (size_t)r * D + c0);
  _Float16* d = tile + r * TS + c0;
#pragma unroll
  for (int q = 0; q < 8; q++) {
    float4 v = p[q];
    v4h h = {(_Float16)v.x, (_Float16)v.y, (_Float16)v.z, (_Float16)v.w};
    *(v4h*)(d + q * 4) = h;
  }
}

DEVINL void coop_copy16(void* dst, const void* src, int n16) {
  uint4* d = (uint4*)dst;
  const uint4* s = (const uint4*)src;
  for (int i = threadIdx.x; i < n16; i += blockDim.x) d[i] = s[i];
}

// ---------------- weight pre-swizzle: W[n][k] -> B-fragment layout (f16) -----
__global__ void k_prep_w(const float* __restrict__ cW1, const float* __restrict__ cW2,
                         const float* __restrict__ oW1, const float* __restrict__ oW2,
                         _Float16* __restrict__ wfr) {
  int tid = blockIdx.x * blockDim.x + threadIdx.x;
  if (tid >= 4 * 4096) return;
  int mat = tid >> 12, r = tid & 4095;
  int j = r & 15, lane = (r >> 4) & 31, kcnb = r >> 9;
  int kc = kcnb & 1, nb = kcnb >> 1;
  int n = nb * 16 + (lane & 15);
  int kb = (lane >> 4) << 3;
  int k = kc * 32 + ((j < 8) ? (kb + j) : (16 + kb + (j - 8)));
  const float* W = (mat == 0) ? cW1 : (mat == 1) ? cW2 : (mat == 2) ? oW1 : oW2;
  wfr[tid] = (_Float16)W[n * D + k];  // B[k][n] = W[n][k] for h @ W^T
}

// ---------------- concat inputs into node buffers ----------------------------
__global__ void k_concat(const float* __restrict__ ue, const float* __restrict__ uo,
                         const float* __restrict__ ie, const float* __restrict__ io,
                         float* __restrict__ embA, float* __restrict__ offA) {
  int i = blockIdx.x * blockDim.x + threadIdx.x;
  if (i >= NN * D) return;
  bool u = i < NUSR * D;
  embA[i] = u ? ue[i] : ie[i - NUSR * D];
  float o = u ? uo[i] : io[i - NUSR * D];
  offA[i] = fmaxf(o, 0.0f);
}

__global__ void k_init_center(unsigned* __restrict__ mkey, float* __restrict__ sbuf,
                              float* __restrict__ wsum) {
  int i = blockIdx.x * blockDim.x + threadIdx.x;
  if (i >= NN * D) return;
  mkey[i] = 0u;  // < key of any finite float; only read where edges exist
  sbuf[i] = 0.0f;
  wsum[i] = 0.0f;
}

__global__ void k_init_off(float* __restrict__ actsum, unsigned* __restrict__ redkey,
                           float* __restrict__ cntf) {
  int i = blockIdx.x * blockDim.x + threadIdx.x;
  if (i >= NSLOT * D) return;
  actsum[i] = 0.0f;
  int slot = i >> 6;
  bool mn = (slot < NUSR) || (slot >= 2 * NUSR + NITM);  // inter & tag use min
  redkey[i] = mn ? 0xFFFFFFFFu : 0u;
  if (i < NSLOT) cntf[i] = 0.0f;
}

// ---------------- center net edge passes (PHASE0 = max, PHASE1 = sums) -------
template <int PHASE>
__global__ __launch_bounds__(256) void k_center_edge(
    const float* __restrict__ emb, const int* __restrict__ head,
    const int* __restrict__ tail, const _Float16* __restrict__ wfr,
    const float* __restrict__ b1, const float* __restrict__ b2,
    unsigned* __restrict__ mkey, float* __restrict__ sbuf,
    float* __restrict__ wsum) {
  __shared__ _Float16 sW[2 * 4096];
  __shared__ _Float16 sA[8][16 * TS];
  __shared__ _Float16 sAct[8][16 * TS];
  coop_copy16(sW, wfr, 1024);
  __syncthreads();
  int lane = threadIdx.x & 31, wid = threadIdx.x >> 5;
  int n = lane & 15, hi8 = (lane >> 4) << 3;
  float bn1[4], bn2[4];
#pragma unroll
  for (int nb = 0; nb < 4; nb++) {
    bn1[nb] = b1[nb * 16 + n];
    bn2[nb] = b2[nb * 16 + n];
  }
  int gw = (gridDim.x * blockDim.x) >> 5;
  int gwid = (blockIdx.x * blockDim.x + threadIdx.x) >> 5;
  for (int tile = gwid; tile < NEDGE / 16; tile += gw) {
    int e0 = tile * 16;
    gather16_idx(sA[wid], emb, tail + e0, lane);
    lds_fence_wave();
    v8f acc[4];
    gemm4(sA[wid], sW, bn1, lane, acc);          // act = relu(h @ cW1^T + cb1)
#pragma unroll
    for (int nb = 0; nb < 4; nb++)
#pragma unroll
      for (int v = 0; v < 8; v++)
        sAct[wid][(hi8 + v) * TS + nb * 16 + n] = (_Float16)fmaxf(acc[nb][v], 0.0f);
    lds_fence_wave();
    gemm4(sAct[wid], sW + 4096, bn2, lane, acc); // logits = act @ cW2^T + cb2
    int hrow[8];
#pragma unroll
    for (int v = 0; v < 8; v++) hrow[v] = head[e0 + hi8 + v];
#pragma unroll
    for (int nb = 0; nb < 4; nb++) {
      int dim = nb * 16 + n;
#pragma unroll
      for (int v = 0; v < 8; v++) {
        size_t idx = (size_t)hrow[v] * D + dim;
        float logit = acc[nb][v];
        if (PHASE == 0) {
          atomicMax(mkey + idx, fkey(logit));    // segment max
        } else {
          float e = __expf(logit - fdecode(mkey[idx]));
          atomicAdd(sbuf + idx, e);              // segment sum of e
          float ev = (float)sA[wid][(hi8 + v) * TS + dim];
          atomicAdd(wsum + idx, e * ev);         // segment sum of e*emb
        }
      }
    }
  }
}

// ---------------- center finalize: agg = wsum/s, row-normalize ---------------
__global__ __launch_bounds__(256) void k_center_fin(const float* __restrict__ sbuf,
                                                    const float* __restrict__ wsum,
                                                    float* __restrict__ embNext) {
  int lane = threadIdx.x & 31;
  int gwid = (blockIdx.x * blockDim.x + threadIdx.x) >> 5;
  int gw = (gridDim.x * blockDim.x) >> 5;
  for (int row = gwid; row < NN; row += gw) {
    size_t b = (size_t)row * D;
    float s0 = sbuf[b + lane], s1 = sbuf[b + lane + 32];
    float v0 = (s0 > 0.0f) ? wsum[b + lane] / s0 : 0.0f;
    float v1 = (s1 > 0.0f) ? wsum[b + lane + 32] / s1 : 0.0f;
    float sq = v0 * v0 + v1 * v1;
#pragma unroll
    for (int o = 16; o > 0; o >>= 1) sq += __shfl_xor(sq, o, 32);
    float sc = 1.0f / fmaxf(sqrtf(sq), 1e-12f);
    embNext[b + lane] = v0 * sc;
    embNext[b + lane + 32] = v1 * sc;
  }
}

// ---------------- offset net edge pass ---------------------------------------
__global__ __launch_bounds__(256) void k_off_edge(
    const float* __restrict__ off, const int* __restrict__ head,
    const int* __restrict__ tail, const _Float16* __restrict__ wfrO1,
    const float* __restrict__ ob1, float* __restrict__ actsum,
    unsigned* __restrict__ redkey, float* __restrict__ cntf) {
  __shared__ _Float16 sW[4096];
  __shared__ _Float16 sT[8][16 * TS];
  coop_copy16(sW, wfrO1, 512);
  __syncthreads();
  int lane = threadIdx.x & 31, wid = threadIdx.x >> 5;
  int n = lane & 15, hi8 = (lane >> 4) << 3;
  float bn1[4];
#pragma unroll
  for (int nb = 0; nb < 4; nb++) bn1[nb] = ob1[nb * 16 + n];
  int gw = (gridDim.x * blockDim.x) >> 5;
  int gwid = (blockIdx.x * blockDim.x + threadIdx.x) >> 5;
  for (int tile = gwid; tile < NEDGE / 16; tile += gw) {
    int e0 = tile * 16;
    gather16_idx(sT[wid], off, tail + e0, lane);
    lds_fence_wave();
    v8f acc[4];
    gemm4(sT[wid], sW, bn1, lane, acc);  // act = relu(edge_off @ oW1^T + ob1)
    int slot8[8], mn8[8];
#pragma unroll
    for (int v = 0; v < 8; v++) {
      int m = hi8 + v;
      int h = head[e0 + m], t = tail[e0 + m];
      int slot = -1, mn = 0;
      if (h < NUSR) {
        if (t >= NUSR) {
          if (t < NUSR + NITM) { slot = h;        mn = 1; }  // inter (min)
          else                 { slot = NUSR + h; mn = 0; }  // ut (max)
        }
      } else {
        slot = NUSR + h;                 // items -> [2NU,2NU+NI), tags after
        mn = (h >= NUSR + NITM) ? 1 : 0; // tag uses min, item uses max
      }
      slot8[v] = slot;
      mn8[v] = mn;
      if (n == 0 && slot >= 0) atomicAdd(cntf + slot, 1.0f);
    }
#pragma unroll
    for (int nb = 0; nb < 4; nb++) {
      int dim = nb * 16 + n;
#pragma unroll
      for (int v = 0; v < 8; v++) {
        if (slot8[v] < 0) continue;
        size_t idx = (size_t)slot8[v] * D + dim;
        atomicAdd(actsum + idx, fmaxf(acc[nb][v], 0.0f));
        float ov = (float)sT[wid][(hi8 + v) * TS + dim];
        unsigned k = fkey(ov);
        if (mn8[v]) atomicMin(redkey + idx, k);
        else        atomicMax(redkey + idx, k);
      }
    }
  }
}

// ---------------- slot finalize: gate = sigmoid(mean @ oW2^T + ob2) ----------
__global__ __launch_bounds__(256) void k_slot_fin(
    const float* __restrict__ actsum, const float* __restrict__ cntf,
    const unsigned* __restrict__ redkey, const _Float16* __restrict__ wfrO2,
    const float* __restrict__ ob2, float* __restrict__ uo,
    float* __restrict__ offNext) {
  __shared__ _Float16 sW[4096];
  __shared__ _Float16 sT[8][16 * TS];
  coop_copy16(sW, wfrO2, 512);
  __syncthreads();
  int lane = threadIdx.x & 31, wid = threadIdx.x >> 5;
  int n = lane & 15, hi8 = (lane >> 4) << 3;
  float bn2[4];
#pragma unroll
  for (int nb = 0; nb < 4; nb++) bn2[nb] = ob2[nb * 16 + n];
  int gw = (gridDim.x * blockDim.x) >> 5;
  int gwid = (blockIdx.x * blockDim.x + threadIdx.x) >> 5;
  for (int tile = gwid; tile < NSLOT / 16; tile += gw) {
    int s0 = tile * 16;
    {  // stage mean rows as f16
      int r = lane >> 1, c0 = (lane & 1) << 5;
      int slot = s0 + r;
      float sc = 1.0f / fmaxf(cntf[slot], 1.0f);
      const float4* p = (const float4*)(actsum + (size_t)slot * D + c0);
      _Float16* dd = sT[wid] + r * TS + c0;
#pragma unroll
      for (int q = 0; q < 8; q++) {
        float4 v = p[q];
        v4h h = {(_Float16)(v.x * sc), (_Float16)(v.y * sc),
                 (_Float16)(v.z * sc), (_Float16)(v.w * sc)};
        *(v4h*)(dd + q * 4) = h;
      }
    }
    lds_fence_wave();
    v8f acc[4];
    gemm4(sT[wid], sW, bn2, lane, acc);
#pragma unroll
    for (int v = 0; v < 8; v++) {
      int slot = s0 + hi8 + v;
      float cn = cntf[slot];
#pragma unroll
      for (int nb = 0; nb < 4; nb++) {
        int dim = nb * 16 + n;
        size_t idx = (size_t)slot * D + dim;
        float g = 1.0f / (1.0f + __expf(-acc[nb][v]));
        float red = (cn > 0.0f) ? fdecode(redkey[idx]) : 0.0f;
        if (!(fabsf(red) <= 3.0e38f)) red = 0.0f;  // empty/inf -> 0
        float o = red * g;
        if (slot < 2 * NUSR) uo[idx] = o;
        else offNext[idx - (size_t)NUSR * D] = o;   // node = slot - NUSR
      }
    }
  }
}

// ---------------- user combine: offset net over [inter_u; ut_u] --------------
__global__ __launch_bounds__(256) void k_user_comb(
    const float* __restrict__ uo, const _Float16* __restrict__ wfrO,
    const float* __restrict__ ob1, const float* __restrict__ ob2,
    float* __restrict__ offNext) {
  __shared__ _Float16 sW[2 * 4096];
  __shared__ _Float16 sT[8][3][16 * TS];
  coop_copy16(sW, wfrO, 1024);
  __syncthreads();
  int lane = threadIdx.x & 31, wid = threadIdx.x >> 5;
  int n = lane & 15, hi8 = (lane >> 4) << 3;
  float bn1[4], bn2[4];
#pragma unroll
  for (int nb = 0; nb < 4; nb++) {
    bn1[nb] = ob1[nb * 16 + n];
    bn2[nb] = ob2[nb * 16 + n];
  }
  int gw = (gridDim.x * blockDim.x) >> 5;
  int gwid = (blockIdx.x * blockDim.x + threadIdx.x) >> 5;
  for (int tile = gwid; tile < NUSR / 16; tile += gw) {
    int u0 = tile * 16;
    gather16_seq(sT[wid][0], uo + (size_t)u0 * D, lane);
    gather16_seq(sT[wid][1], uo + (size_t)(NUSR + u0) * D, lane);
    lds_fence_wave();
    v8f accA[4], accB[4];
    gemm4(sT[wid][0], sW, bn1, lane, accA);
    gemm4(sT[wid][1], sW, bn1, lane, accB);
#pragma unroll
    for (int nb = 0; nb < 4; nb++)
#pragma unroll
      for (int v = 0; v < 8; v++) {
        float mean = (fmaxf(accA[nb][v], 0.0f) + fmaxf(accB[nb][v], 0.0f)) * 0.5f;
        sT[wid][2][(hi8 + v) * TS + nb * 16 + n] = (_Float16)mean;
      }
    lds_fence_wave();
    v8f acc[4];
    gemm4(sT[wid][2], sW + 4096, bn2, lane, acc);
#pragma unroll
    for (int v = 0; v < 8; v++) {
      int m = hi8 + v;
#pragma unroll
      for (int nb = 0; nb < 4; nb++) {
        int dim = nb * 16 + n;
        float g = 1.0f / (1.0f + __expf(-acc[nb][v]));
        float red = fmaxf((float)sT[wid][0][m * TS + dim],
                          (float)sT[wid][1][m * TS + dim]);
        offNext[(size_t)(u0 + m) * D + dim] = fmaxf(red * g, 0.0f);
      }
    }
  }
}

// ---------------- final packing ----------------------------------------------
__global__ void k_copy_out(const float* __restrict__ embF,
                           const float* __restrict__ offF,
                           float* __restrict__ out) {
  int i = blockIdx.x * blockDim.x + threadIdx.x;
  constexpr int A = NUSR * D, B = 2 * NUSR * D, C = (2 * NUSR + NENT) * D;
  constexpr int TOT = (2 * NUSR + 2 * NENT) * D;
  if (i >= TOT) return;
  float v;
  if (i < A)      v = embF[i];
  else if (i < B) v = offF[i - A];
  else if (i < C) v = embF[i - A];
  else            v = offF[i - (NUSR + NENT) * D];
  out[i] = v;
}

// =============================================================================
extern "C" void kernel_launch(void* const* d_in, const int* in_sizes, int n_in,
                              void* d_out, int out_size, void* d_ws,
                              size_t ws_size, hipStream_t stream) {
  (void)in_sizes; (void)n_in; (void)out_size; (void)ws_size;
  const float* user_emb = (const float*)d_in[0];
  const float* user_off = (const float*)d_in[1];
  const float* item_emb = (const float*)d_in[2];
  const float* item_off = (const float*)d_in[3];
  const float* cW1 = (const float*)d_in[4];
  const float* cb1 = (const float*)d_in[5];
  const float* cW2 = (const float*)d_in[6];
  const float* cb2 = (const float*)d_in[7];
  const float* oW1 = (const float*)d_in[8];
  const float* ob1 = (const float*)d_in[9];
  const float* oW2 = (const float*)d_in[10];
  const float* ob2 = (const float*)d_in[11];
  const int* head = (const int*)d_in[12];
  const int* tail = (const int*)d_in[13];
  float* out = (float*)d_out;

  // workspace layout (float units)
  float* base = (float*)d_ws;
  size_t o = 0;
  float* embA = base + o;   o += (size_t)NN * D;
  float* embB = base + o;   o += (size_t)NN * D;
  float* offA = base + o;   o += (size_t)NN * D;
  float* offB = base + o;   o += (size_t)NN * D;
  float* sbuf = base + o;   o += (size_t)NN * D;
  float* wsum = base + o;   o += (size_t)NN * D;
  unsigned* mkey = (unsigned*)(base + o);   o += (size_t)NN * D;
  float* actsum = base + o; o += (size_t)NSLOT * D;
  unsigned* redkey = (unsigned*)(base + o); o += (size_t)NSLOT * D;
  float* cntf = base + o;   o += (size_t)NSLOT;
  float* uo = base + o;     o += (size_t)2 * NUSR * D;
  _Float16* wfr = (_Float16*)(base + o);    // 4*4096 halves = 8192 floats

  const int T = 256;
  k_prep_w<<<(4 * 4096 + T - 1) / T, T, 0, stream>>>(cW1, cW2, oW1, oW2, wfr);
  k_concat<<<(NN * D + T - 1) / T, T, 0, stream>>>(user_emb, user_off, item_emb,
                                                   item_off, embA, offA);

  float* ce = embA; float* co = offA;
  float* ne = embB; float* no = offB;
  for (int L = 0; L < 2; L++) {
    k_init_center<<<(NN * D + T - 1) / T, T, 0, stream>>>(mkey, sbuf, wsum);
    k_init_off<<<(NSLOT * D + T - 1) / T, T, 0, stream>>>(actsum, redkey, cntf);

    k_center_edge<0><<<500, T, 0, stream>>>(ce, head, tail, wfr, cb1, cb2, mkey,
                                            sbuf, wsum);
    k_center_edge<1><<<500, T, 0, stream>>>(ce, head, tail, wfr, cb1, cb2, mkey,
                                            sbuf, wsum);
    k_center_fin<<<(NN + 7) / 8, T, 0, stream>>>(sbuf, wsum, ne);

    k_off_edge<<<500, T, 0, stream>>>(co, head, tail, wfr + 2 * 4096, ob1,
                                      actsum, redkey, cntf);
    k_slot_fin<<<400, T, 0, stream>>>(actsum, cntf, redkey, wfr + 3 * 4096, ob2,
                                      uo, no);
    k_user_comb<<<200, T, 0, stream>>>(uo, wfr + 2 * 4096, ob1, ob2, no);

    float* te = ce; ce = ne; ne = te;
    float* to = co; co = no; no = to;
  }
  k_copy_out<<<((2 * NUSR + 2 * NENT) * D + T - 1) / T, T, 0, stream>>>(ce, co,
                                                                        out);
}
